// MultiLoraModule_45956150067888
// MI455X (gfx1250) — matrix operand order repaired and checked by reading, not compiled, not run on hardware
//
#include <hip/hip_runtime.h>

#define B_    4
#define S_    2048
#define DIN   4096
#define DOUT  4096
#define RANK  16

#define BM    128
#define BN    128
#define KC    64          // K elements per LDS stage (2 WMMA k-steps)
#define STR   72          // LDS row stride in bf16 (144B: 16B-aligned, bank-spread)

typedef __attribute__((ext_vector_type(16))) __bf16 v16bf;
typedef __attribute__((ext_vector_type(8)))  __bf16 v8bf;
typedef __attribute__((ext_vector_type(8)))  float  v8f;

#define AS3 __attribute__((address_space(3)))

#if __has_builtin(__builtin_amdgcn_global_load_async_to_lds_b128)
#define HAVE_ASYNC_LDS 1      // builtin presence == gfx1250 async-LDS support
#else
#define HAVE_ASYNC_LDS 0
#endif

union BF16x16 {
    v16bf v;
    v8bf  h[2];
    __bf16 e[16];
    uint4  u[2];
};

__device__ __forceinline__ void cvt8(float4 a, float4 b, __bf16* e) {
    e[0] = (__bf16)a.x; e[1] = (__bf16)a.y; e[2] = (__bf16)a.z; e[3] = (__bf16)a.w;
    e[4] = (__bf16)b.x; e[5] = (__bf16)b.y; e[6] = (__bf16)b.z; e[7] = (__bf16)b.w;
}

__device__ __forceinline__ v16bf join8(v8bf lo, v8bf hi) {
    BF16x16 u; u.h[0] = lo; u.h[1] = hi; return u.v;
}

// 16B global -> LDS copy. Async path (ASYNCcnt-tracked, no VGPR round trip)
// via inline asm per CDNA5 ISA VGLOBAL async encoding:
//   global_load_async_to_lds_b128 vDstLdsAddr, v[gaddr64], off
__device__ __forceinline__ void copy_b128_to_lds(const __bf16* g, __bf16* l) {
#if HAVE_ASYNC_LDS
    unsigned lds_off = (unsigned)(size_t)(AS3 __bf16*)l;   // 32-bit LDS byte addr
    unsigned long long ga = (unsigned long long)(size_t)g;
    asm volatile("global_load_async_to_lds_b128 %0, %1, off"
                 :: "v"(lds_off), "v"(ga)
                 : "memory");
#else
    *(uint4*)l = *(const uint4*)g;
#endif
}

__device__ __forceinline__ void wait_stage() {
#if HAVE_ASYNC_LDS
 #if __has_builtin(__builtin_amdgcn_s_wait_asynccnt)
    __builtin_amdgcn_s_wait_asynccnt(0);
 #else
    asm volatile("s_wait_asynccnt 0" ::: "memory");
 #endif
#endif
}

// ---------------------------------------------------------------------------
// Kernel 1: generic f32 -> bf16 conversion (used for base_weight and x).
// ---------------------------------------------------------------------------
__global__ __launch_bounds__(256) void k_conv(const float* __restrict__ src,
                                              __bf16* __restrict__ dst) {
    size_t i = ((size_t)blockIdx.x * 256 + threadIdx.x) * 8;
    const float4* p = (const float4*)(src + i);
    float4 a = p[0], b = p[1];
    BF16x16 o;
    cvt8(a, b, o.e);
    *(uint4*)(dst + i) = o.u[0];
}

// ---------------------------------------------------------------------------
// Kernel 2: gather lora_B_w[aid[b]] -> bf16, pad RANK 16 -> 32 with zeros.
// bwp layout: [B][DOUT][32] bf16
// ---------------------------------------------------------------------------
__global__ __launch_bounds__(256) void k_bw_pad(const float* __restrict__ loraB,
                                                const int* __restrict__ aid,
                                                __bf16* __restrict__ bwp) {
    int t = blockIdx.x * 256 + threadIdx.x;     // [0, B_*DOUT)
    int b = t / DOUT, o = t % DOUT;
    int a = aid[b];
    const float* src = loraB + ((size_t)a * DOUT + o) * RANK;
    union { __bf16 e[32]; uint4 u[4]; } row;
    #pragma unroll
    for (int r = 0; r < 16; ++r) row.e[r] = (__bf16)src[r];
    #pragma unroll
    for (int r = 16; r < 32; ++r) row.e[r] = (__bf16)0.f;
    uint4* d = (uint4*)(bwp + (size_t)t * 32);
    d[0] = row.u[0]; d[1] = row.u[1]; d[2] = row.u[2]; d[3] = row.u[3];
}

// ---------------------------------------------------------------------------
// Kernel 3: h = x @ A[aid]^T via WMMA. One wave owns a 16-row tile, RANK=16=N.
// hp layout: [B][S][32] bf16 (upper 16 zero-padded)
// ---------------------------------------------------------------------------
__global__ __launch_bounds__(256) void k_h(const __bf16* __restrict__ xb16,
                                           const float* __restrict__ loraA,
                                           const int* __restrict__ aid,
                                           __bf16* __restrict__ hp) {
    int wave = (blockIdx.x * 256 + threadIdx.x) >> 5;
    int lane = threadIdx.x & 31;
    int b  = wave / (S_ / 16);
    int st = wave % (S_ / 16);
    int a  = aid[b];

    const __bf16* xr = xb16 + ((size_t)b * S_   + st * 16 + (lane & 15)) * DIN;
    const float*  ar = loraA + ((size_t)a * RANK +          (lane & 15)) * DIN;
    int kbA = (lane >> 4) * 8;     // A fragment: K[kb..kb+7] | K[kb+16..kb+23]
    int kbB = (lane >> 4) * 8;     // B fragment: K[kb..kb+7] | K[kb+16..kb+23]

    v8f c = {};
    for (int kc = 0; kc < DIN / 32; ++kc) {
        int k0 = kc * 32;
        v16bf af = join8(*(const v8bf*)(xr + k0 + kbA),
                         *(const v8bf*)(xr + k0 + kbA + 16));
        BF16x16 bf;
        const float4* pb = (const float4*)(ar + k0 + kbB);
        cvt8(pb[0], pb[1], bf.e);
        cvt8(pb[4], pb[5], bf.e + 8);
        c = __builtin_amdgcn_wmma_f32_16x16x32_bf16(false, af, false, bf.v,
                                                    (short)0, c, false, false);
    }

    __bf16* hrow = hp + ((size_t)b * S_ + st * 16) * 32;
    int r   = lane & 15;
    int mhi = (lane >> 4) * 8;
    #pragma unroll
    for (int i = 0; i < 8; ++i) {
        int m = i + mhi;
        hrow[(size_t)m * 32 + r]      = (__bf16)c[i];
        hrow[(size_t)m * 32 + r + 16] = (__bf16)0.f;
    }
}

// ---------------------------------------------------------------------------
// Kernel 4: main fused GEMM (all-bf16 operands, f32 accumulate).
// out[b,m,o] = sum_k x[b,m,k]*W[o,k] + bias[o] + sum_r h[b,m,r]*Bw[b,o,r]
// K loop: DIN/64 double-buffered LDS stages (2 WMMA k-steps each) + 1 fused
// LoRA stage (zero-padded h / B_pad). 128x128 block tile, 8 waves of 64x32.
// ---------------------------------------------------------------------------
__global__ __launch_bounds__(256) void k_main(const __bf16* __restrict__ xb16,
                                              const __bf16* __restrict__ wb16,
                                              const float*  __restrict__ bias,
                                              const __bf16* __restrict__ hp,
                                              const __bf16* __restrict__ bwp,
                                              float* __restrict__ out) {
    __shared__ __bf16 lx[2][BM][STR];
    __shared__ __bf16 lw[2][BN][STR];

    const int tid  = threadIdx.x;
    const int lane = tid & 31;
    const int wave = tid >> 5;
    const int wm = wave >> 2;        // 0..1 -> 64-row slab
    const int wn = wave & 3;         // 0..3 -> 32-col slab

    const int nb_n = DOUT / BN;      // 32
    const int nb_m = S_ / BM;        // 16
    const int bn = blockIdx.x % nb_n;
    const int bm = (blockIdx.x / nb_n) % nb_m;
    const int b  = blockIdx.x / (nb_n * nb_m);

    const __bf16* xb  = xb16 + (size_t)b * S_ * DIN;
    const __bf16* hb  = hp   + (size_t)b * S_ * 32;
    const __bf16* bwb = bwp  + (size_t)b * DOUT * 32;

    // staging: thread t -> row t/2, 32-halves slab (t&1) for K=64 stages
    const int    srow  = tid >> 1;
    const int    scol  = (tid & 1) * 32;
    const int    scol2 = (tid & 1) * 16;     // for the 32-wide LoRA stage
    const size_t gmrow = (size_t)bm * BM + srow;
    const size_t gorow = (size_t)bn * BN + srow;

    const int NST = DIN / KC;        // 64 normal stages; stage NST == LoRA

    auto stage = [&](int kc, int buf) {
        if (kc < NST) {
            const __bf16* gx = xb   + gmrow * DIN + kc * KC + scol;
            const __bf16* gw = wb16 + gorow * DIN + kc * KC + scol;
            __bf16* dx = &lx[buf][srow][scol];
            __bf16* dw = &lw[buf][srow][scol];
            #pragma unroll
            for (int j = 0; j < 4; ++j) copy_b128_to_lds(gx + j * 8, dx + j * 8);
            #pragma unroll
            for (int j = 0; j < 4; ++j) copy_b128_to_lds(gw + j * 8, dw + j * 8);
        } else {
            const __bf16* gx = hb  + gmrow * 32 + scol2;
            const __bf16* gw = bwb + gorow * 32 + scol2;
            __bf16* dx = &lx[buf][srow][scol2];
            __bf16* dw = &lw[buf][srow][scol2];
            #pragma unroll
            for (int j = 0; j < 2; ++j) copy_b128_to_lds(gx + j * 8, dx + j * 8);
            #pragma unroll
            for (int j = 0; j < 2; ++j) copy_b128_to_lds(gw + j * 8, dw + j * 8);
        }
    };

    v8f acc[4][2] = {};

    stage(0, 0);
    wait_stage();
    __syncthreads();

    const int mrow = lane & 15;
    const int kbA  = (lane >> 4) * 8;
    const int kbB  = (lane >> 4) * 8;

    auto mma_step = [&](int cur, int ksbase) {
        v16bf afr[4], bfr[2];
        #pragma unroll
        for (int mt = 0; mt < 4; ++mt) {
            int row = wm * 64 + mt * 16 + mrow;
            afr[mt] = join8(*(const v8bf*)&lx[cur][row][ksbase + kbA],
                            *(const v8bf*)&lx[cur][row][ksbase + kbA + 16]);
        }
        #pragma unroll
        for (int nt = 0; nt < 2; ++nt) {
            int row = wn * 32 + nt * 16 + mrow;
            bfr[nt] = join8(*(const v8bf*)&lw[cur][row][ksbase + kbB],
                            *(const v8bf*)&lw[cur][row][ksbase + kbB + 16]);
        }
        #pragma unroll
        for (int mt = 0; mt < 4; ++mt)
            #pragma unroll
            for (int nt = 0; nt < 2; ++nt)
                acc[mt][nt] = __builtin_amdgcn_wmma_f32_16x16x32_bf16(
                    false, afr[mt], false, bfr[nt], (short)0, acc[mt][nt],
                    false, false);
    };

    for (int kc = 0; kc <= NST; ++kc) {
        const int cur = kc & 1;
        if (kc < NST) stage(kc + 1, cur ^ 1);   // async into other buffer
        mma_step(cur, 0);
        if (kc < NST) {
            mma_step(cur, 32);
            wait_stage();                        // my async writes landed in LDS
            __syncthreads();                     // everyone's landed
        }
    }

    // epilogue: + bias, store f32
    float* ob = out + (size_t)b * S_ * DOUT;
    const int nloc = lane & 15;
    const int mhi  = (lane >> 4) * 8;
    #pragma unroll
    for (int nt = 0; nt < 2; ++nt) {
        int o = bn * BN + wn * 32 + nt * 16 + nloc;
        float bv = bias[o];
        #pragma unroll
        for (int mt = 0; mt < 4; ++mt) {
            int mbase = bm * BM + wm * 64 + mt * 16 + mhi;
            #pragma unroll
            for (int i = 0; i < 8; ++i)
                ob[(size_t)(mbase + i) * DOUT + o] = acc[mt][nt][i] + bv;
        }
    }
}

// ---------------------------------------------------------------------------
extern "C" void kernel_launch(void* const* d_in, const int* in_sizes, int n_in,
                              void* d_out, int out_size, void* d_ws, size_t ws_size,
                              hipStream_t stream) {
    (void)in_sizes; (void)n_in; (void)out_size; (void)ws_size;
    const float* x    = (const float*)d_in[0];
    const int*   aid  = (const int*)  d_in[1];
    const float* w    = (const float*)d_in[2];
    const float* bias = (const float*)d_in[3];
    const float* lA   = (const float*)d_in[4];
    const float* lB   = (const float*)d_in[5];
    float* out = (float*)d_out;

    char* ws = (char*)d_ws;
    size_t off = 0;
    __bf16* wb  = (__bf16*)(ws + off); off += (size_t)DOUT * DIN * 2;      // 32 MB
    __bf16* xb  = (__bf16*)(ws + off); off += (size_t)B_ * S_ * DIN * 2;   // 64 MB
    __bf16* hp  = (__bf16*)(ws + off); off += (size_t)B_ * S_ * 32 * 2;    // 512 KB
    __bf16* bwp = (__bf16*)(ws + off);                                     // 1 MB

    k_conv<<<(int)((size_t)DOUT * DIN / 2048), 256, 0, stream>>>(w, wb);
    k_conv<<<(int)((size_t)B_ * S_ * DIN / 2048), 256, 0, stream>>>(x, xb);
    k_bw_pad<<<(B_ * DOUT) / 256, 256, 0, stream>>>(lB, aid, bwp);
    k_h<<<(B_ * (S_ / 16)) / 8, 256, 0, stream>>>(xb, lA, aid, hp);
    k_main<<<B_ * (S_ / BM) * (DOUT / BN), 256, 0, stream>>>(xb, wb, bias, hp, bwp, out);
}
